// GATBase_21225728377482
// MI455X (gfx1250) — compile-verified, matrix operand order checked by reading, not run on hardware
//
#include <hip/hip_runtime.h>
#include <hip/hip_bf16.h>

typedef __attribute__((ext_vector_type(16))) _Float16 v16h;
typedef __attribute__((ext_vector_type(8)))  float    v8f;
typedef __attribute__((ext_vector_type(4)))  unsigned v4u;
typedef __attribute__((ext_vector_type(8)))  int      v8i;
typedef __attribute__((ext_vector_type(4)))  int      v4i;
typedef _Float16 half_t;

#define N_D 64
#define DZ 192          // 3*D
#define NEG_SLOPE 0.2f
#define EU_WAVES 2
#define SNH 136         // padded half stride for znode rows (128 + 8)
#define SZE 68          // padded float stride for zedge rows (64 + 4), matches TDM pad
#define SZF 200         // padded float stride for h1 rows
#define SZH 200         // padded half stride for h1n rows

// per-wave LDS region layout (bytes), all 16B aligned
#define OFF_ZNODE 0
#define OFF_ZEDGE 4352              // 16*SNH*2  = 4352
#define OFF_H1    8704              // + 16*SZE*4 = 4352
#define OFF_H1N   21504             // + 16*SZF*4 = 12800
#define WAVE_LDS  27904             // + 16*SZH*2 = 6400

// ---------- helpers ----------
__device__ __forceinline__ unsigned mapf(float f) {
    unsigned u = __float_as_uint(f);
    return (u & 0x80000000u) ? ~u : (u | 0x80000000u);
}
__device__ __forceinline__ float unmapf(unsigned m) {
    unsigned u = (m & 0x80000000u) ? (m & 0x7FFFFFFFu) : ~m;
    return __uint_as_float(u);
}
__device__ __forceinline__ float lrelu(float x) { return x > 0.f ? x : NEG_SLOPE * x; }

// ---------- tiny per-layer precompute: vs/vd/ve = W @ att (64x4 each) ----------
__global__ void prep_vecs_kernel(const float* __restrict__ W, const float* __restrict__ We,
                                 const float* __restrict__ as_, const float* __restrict__ ad_,
                                 const float* __restrict__ ae_,
                                 float* __restrict__ vs, float* __restrict__ vd, float* __restrict__ ve) {
    int t = threadIdx.x;           // 256 = 64 * 4
    int d = t >> 2, h = t & 3;
    float s1 = 0.f, s2 = 0.f, s3 = 0.f;
    #pragma unroll
    for (int c = 0; c < 16; ++c) {
        float w  = W[d * 64 + h * 16 + c];
        float we = We[d * 64 + h * 16 + c];
        s1 += w  * as_[h * 16 + c];
        s2 += w  * ad_[h * 16 + c];
        s3 += we * ae_[h * 16 + c];
    }
    vs[d * 4 + h] = s1; vd[d * 4 + h] = s2; ve[d * 4 + h] = s3;
}

// ---------- pack fp32 weight [K,N] into WMMA B-fragment-ordered f16 ----------
__global__ void frag_pack_kernel(const float* __restrict__ w, half_t* __restrict__ out,
                                 int K, int N) {
    int NT = N >> 4;
    int total = K * N;
    int t = blockIdx.x * blockDim.x + threadIdx.x;
    if (t >= total) return;
    int i    = t & 15;
    int lane = (t >> 4) & 31;
    int blk  = t >> 9;             // kc*NT + nt
    int nt   = blk % NT;
    int kc   = blk / NT;
    int n     = nt * 16 + (lane & 15);
    int khalf = (lane >> 4) * 8;
    int v = i >> 1, hh = i & 1;
    int k = kc * 32 + khalf + ((v < 4) ? (2 * v + hh) : (16 + 2 * (v - 4) + hh));
    out[t] = (half_t)w[(long)k * N + n];
}

// ---------- node linear xh = nf @ W (fp32, thread = (node, 4-col group)) ----------
__global__ void node_xh_kernel(const float* __restrict__ nf, const float* __restrict__ W,
                               float* __restrict__ xh, int N) {
    int t = blockIdx.x * blockDim.x + threadIdx.x;
    if (t >= N * 16) return;
    int node = t >> 4, j4 = (t & 15) * 4;
    const float* row = nf + (long)node * 64;
    float a0 = 0.f, a1 = 0.f, a2 = 0.f, a3 = 0.f;
    #pragma unroll 8
    for (int d = 0; d < 64; ++d) {
        float x = row[d];
        const float* wr = W + d * 64 + j4;
        a0 += x * wr[0]; a1 += x * wr[1]; a2 += x * wr[2]; a3 += x * wr[3];
    }
    float4 o; o.x = a0; o.y = a1; o.z = a2; o.w = a3;
    *(float4*)(xh + (long)node * 64 + j4) = o;
}

// ---------- per-node attention scalars + self-loop alpha + amax init ----------
__global__ void node_attn_kernel(const float* __restrict__ nf,
                                 const float* __restrict__ vs, const float* __restrict__ vd,
                                 float* __restrict__ a_src, float* __restrict__ a_dst,
                                 float* __restrict__ alpha_self, unsigned* __restrict__ amax,
                                 int N) {
    int i = blockIdx.x * blockDim.x + threadIdx.x;
    if (i >= N) return;
    const float* row = nf + (long)i * 64;
    float s[4] = {0.f, 0.f, 0.f, 0.f}, dd[4] = {0.f, 0.f, 0.f, 0.f};
    for (int d = 0; d < 64; ++d) {
        float x = row[d];
        #pragma unroll
        for (int h = 0; h < 4; ++h) { s[h] += x * vs[d * 4 + h]; dd[h] += x * vd[d * 4 + h]; }
    }
    #pragma unroll
    for (int h = 0; h < 4; ++h) {
        a_src[i * 4 + h] = s[h];
        a_dst[i * 4 + h] = dd[h];
        float al = lrelu(s[h] + dd[h]);      // self loop: a_edge = 0
        alpha_self[i * 4 + h] = al;
        amax[i * 4 + h] = mapf(al);
    }
}

// ---------- per-edge alpha + segment max (monotonic-uint atomicMax) ----------
__global__ void edge_alpha_kernel(const float* __restrict__ ef, const float* __restrict__ ve,
                                  const float* __restrict__ a_src, const float* __restrict__ a_dst,
                                  const int* __restrict__ es, const int* __restrict__ ed,
                                  float* __restrict__ alpha, unsigned* __restrict__ amax, int E) {
    int e = blockIdx.x * blockDim.x + threadIdx.x;
    if (e >= E) return;
    const float* row = ef + (long)e * 64;
    float a[4] = {0.f, 0.f, 0.f, 0.f};
    for (int d = 0; d < 64; ++d) {
        float x = row[d];
        #pragma unroll
        for (int h = 0; h < 4; ++h) a[h] += x * ve[d * 4 + h];
    }
    int s = es[e], d2 = ed[e];
    #pragma unroll
    for (int h = 0; h < 4; ++h) {
        float al = lrelu(a_src[s * 4 + h] + a_dst[d2 * 4 + h] + a[h]);
        alpha[(long)e * 4 + h] = al;
        atomicMax(&amax[d2 * 4 + h], mapf(al));
    }
}

// ---------- node: ex_self, denom init (non-atomic, single writer) ----------
__global__ void node_denom_kernel(const unsigned* __restrict__ amax,
                                  float* __restrict__ alpha_self_ex,
                                  float* __restrict__ denom, int N) {
    int i = blockIdx.x * blockDim.x + threadIdx.x;
    if (i >= N) return;
    #pragma unroll
    for (int h = 0; h < 4; ++h) {
        float am = unmapf(amax[i * 4 + h]);
        float ex = __expf(alpha_self_ex[i * 4 + h] - am);
        alpha_self_ex[i * 4 + h] = ex;
        denom[i * 4 + h] = ex;
    }
}

// ---------- edge: ex + denom accumulation ----------
__global__ void edge_exp_kernel(const int* __restrict__ ed, const unsigned* __restrict__ amax,
                                float* __restrict__ alpha_ex, float* __restrict__ denom, int E) {
    int e = blockIdx.x * blockDim.x + threadIdx.x;
    if (e >= E) return;
    int d2 = ed[e];
    #pragma unroll
    for (int h = 0; h < 4; ++h) {
        float am = unmapf(amax[d2 * 4 + h]);
        float ex = __expf(alpha_ex[(long)e * 4 + h] - am);
        alpha_ex[(long)e * 4 + h] = ex;
        atomicAdd(&denom[d2 * 4 + h], ex);
    }
}

// ---------- node: init output accumulator with self-loop message ----------
__global__ void node_out_init_kernel(const float* __restrict__ xh, const float* __restrict__ ex_self,
                                     const float* __restrict__ denom, float* __restrict__ outb, int N) {
    int i = blockIdx.x * blockDim.x + threadIdx.x;
    if (i >= N) return;
    float coef[4];
    #pragma unroll
    for (int h = 0; h < 4; ++h) coef[h] = ex_self[i * 4 + h] / (denom[i * 4 + h] + 1e-16f);
    const float* xr = xh + (long)i * 64;
    float* orow = outb + (long)i * 64;
    #pragma unroll
    for (int c = 0; c < 64; ++c) orow[c] = xr[c] * coef[c >> 4];
}

// ---------- edge: scatter messages (thread = edge*4 + head, 16 atomics) ----------
__global__ void edge_scatter_kernel(const int* __restrict__ es, const int* __restrict__ ed,
                                    const float* __restrict__ xh, const float* __restrict__ ex,
                                    const float* __restrict__ denom, float* __restrict__ outb, int E) {
    int t = blockIdx.x * blockDim.x + threadIdx.x;
    if (t >= E * 4) return;
    int e = t >> 2, h = t & 3;
    int d2 = ed[e], s = es[e];
    float coef = ex[(long)e * 4 + h] / (denom[d2 * 4 + h] + 1e-16f);
    const float* xs = xh + (long)s * 64 + h * 16;
    float* op = outb + (long)d2 * 64 + h * 16;
    #pragma unroll
    for (int c = 0; c < 16; ++c) atomicAdd(&op[c], xs[c] * coef);
}

// ---------- node finalize: bias + LN + (relu) + residual ----------
__global__ void node_finalize_kernel(const float* __restrict__ outb, const float* __restrict__ bias,
                                     const float* __restrict__ lnw, const float* __restrict__ lnb,
                                     const float* __restrict__ nf_in, float* __restrict__ nf_out,
                                     int N, int do_relu) {
    int i = blockIdx.x * blockDim.x + threadIdx.x;
    if (i >= N) return;
    float v[64];
    const float* r = outb + (long)i * 64;
    float s = 0.f;
    #pragma unroll
    for (int c = 0; c < 64; ++c) { v[c] = r[c] + bias[c]; s += v[c]; }
    float mu = s * (1.f / 64.f);
    float s2 = 0.f;
    #pragma unroll
    for (int c = 0; c < 64; ++c) { float d = v[c] - mu; s2 += d * d; }
    float rs = rsqrtf(s2 * (1.f / 64.f) + 1e-5f);
    const float* nin = nf_in + (long)i * 64;
    float* nout = nf_out + (long)i * 64;
    #pragma unroll
    for (int c = 0; c < 64; ++c) {
        float x = (v[c] - mu) * rs * lnw[c] + lnb[c];
        if (do_relu) x = x > 0.f ? x : 0.f;
        nout[c] = x + nin[c];
    }
}

// ---------- WMMA fragment loaders ----------
// GEMM1 A operand: cols 0..127 from znode (f16), cols 128..191 from TDM-staged zedge (f32)
__device__ __forceinline__ v16h load_afrag_z(const half_t* zn, const float* ze, int lane, int kc) {
    int m = lane & 15;
    int khalf = (lane >> 4) * 8;
    v16h a;
    if (kc < 4) {
        const half_t* r = zn + m * SNH + kc * 32 + khalf;
        #pragma unroll
        for (int v = 0; v < 8; ++v) {
            int k = (v < 4) ? (2 * v) : (16 + 2 * (v - 4));
            a[2 * v]     = r[k];
            a[2 * v + 1] = r[k + 1];
        }
    } else {
        const float* r = ze + m * SZE + (kc - 4) * 32 + khalf;
        #pragma unroll
        for (int v = 0; v < 8; ++v) {
            int k = (v < 4) ? (2 * v) : (16 + 2 * (v - 4));
            a[2 * v]     = (half_t)r[k];
            a[2 * v + 1] = (half_t)r[k + 1];
        }
    }
    return a;
}
__device__ __forceinline__ v16h load_afrag_h(const half_t* base, int lane, int kc) {
    int m = lane & 15;
    int khalf = (lane >> 4) * 8;
    const half_t* r = base + m * SZH + kc * 32 + khalf;
    v16h a;
    #pragma unroll
    for (int v = 0; v < 8; ++v) {
        int k = (v < 4) ? (2 * v) : (16 + 2 * (v - 4));
        a[2 * v]     = r[k];
        a[2 * v + 1] = r[k + 1];
    }
    return a;
}
__device__ __forceinline__ v16h load_bfrag(const half_t* __restrict__ w, int lane, int kc,
                                           int nt, int NT) {
    const half_t* p = w + (((long)kc * NT + nt) * 32 + lane) * 16;   // 32B aligned, contiguous
    return *(const v16h*)p;
}

// ---------- fused edge-update MLP: (gather + TDM) -> GEMM1 -> LN+ReLU -> GEMM2 -> residual ----
__global__ __launch_bounds__(EU_WAVES * 32)
void edge_update_kernel(const float* __restrict__ nf,
                        const float* __restrict__ ef_in, float* __restrict__ ef_out,
                        const int* __restrict__ es, const int* __restrict__ ed,
                        const half_t* __restrict__ w1f, const float* __restrict__ b1,
                        const float* __restrict__ lnw, const float* __restrict__ lnb,
                        const half_t* __restrict__ w2f, const float* __restrict__ b2,
                        int E) {
    __shared__ __align__(16) char smem[EU_WAVES * WAVE_LDS];
    int wave = threadIdx.x >> 5;
    int lane = threadIdx.x & 31;
    long e0 = ((long)blockIdx.x * EU_WAVES + wave) * 16;

    half_t* znode = (half_t*)(smem + wave * WAVE_LDS + OFF_ZNODE);
    float*  zedge = (float*)(smem + wave * WAVE_LDS + OFF_ZEDGE);
    float*  h1    = (float*)(smem + wave * WAVE_LDS + OFF_H1);
    half_t* h1n   = (half_t*)(smem + wave * WAVE_LDS + OFF_H1N);

    // ---- TDM: async-load contiguous 16x64 fp32 edge-feature tile -> zedge (stride 68 dw) ----
    if (e0 < (long)E) {
        unsigned long ga = (unsigned long)(const void*)(ef_in + e0 * 64);
        unsigned dim1 = (unsigned)((long)E - e0);        // remaining rows; OOB rows -> zeros
        v4u g0;
        g0[0] = 1u;                                      // count=1, user descriptor
        g0[1] = (unsigned)(wave * WAVE_LDS + OFF_ZEDGE); // lds_addr (smem starts at LDS 0)
        g0[2] = (unsigned)(ga & 0xFFFFFFFFu);            // global_addr lo
        g0[3] = (unsigned)((ga >> 32) & 0x01FFFFFFu) | (2u << 30);  // global_addr hi | type=2
        v8i g1;
        g1[0] = (int)((2u << 16) | (1u << 20) | (5u << 22) | (3u << 25));
                 // data_size=4B | pad_enable | pad_interval=64dw | pad_amount=4dw
        g1[1] = (int)(64u << 16);                        // tensor_dim0 = 64 (lo16)
        g1[2] = (int)((dim1 & 0xFFFFu) << 16);           // tensor_dim1 lo16
        g1[3] = (int)((dim1 >> 16) | (64u << 16));       // tensor_dim1 hi16 | tile_dim0=64
        g1[4] = 16;                                      // tile_dim1=16, tile_dim2=0
        g1[5] = 64;                                      // tensor_dim0_stride = 64
        g1[6] = (int)(64u << 16);                        // tensor_dim1_stride lo16 = 64
        g1[7] = 0;
        v4i gz4 = {0, 0, 0, 0};
        v8i gz8 = {0, 0, 0, 0, 0, 0, 0, 0};
        // 6-arg form (clang-23 / therock-10.0 headers): groups 2/3 unused for this 2D tile
        __builtin_amdgcn_tensor_load_to_lds(g0, g1, gz4, gz4, gz8, 0);
    }

    // ---- per-lane gather: [nf[src] | nf[dst]] as f16 -> znode (512 float4 / wave) ----
    #pragma unroll
    for (int it = 0; it < 16; ++it) {
        int f = it * 32 + lane;         // 0..511
        int row = f >> 5;
        int seg = f & 31;
        long e = e0 + row;
        long ec = e < E ? e : (long)(E - 1);
        const float* srcp = (seg < 16) ? (nf + (long)es[ec] * 64 + seg * 4)
                                       : (nf + (long)ed[ec] * 64 + (seg - 16) * 4);
        float4 v = *(const float4*)srcp;
        half_t* dst = znode + row * SNH + seg * 4;
        dst[0] = (half_t)v.x; dst[1] = (half_t)v.y; dst[2] = (half_t)v.z; dst[3] = (half_t)v.w;
    }
    __syncthreads();
    __builtin_amdgcn_s_wait_tensorcnt(0);   // zedge ready (same-wave consumer)

    // ---- GEMM1: h1 = z @ w1 + b1   (16x192 @ 192x192, f16 in, f32 acc) ----
    v16h afr[6];
    #pragma unroll
    for (int kc = 0; kc < 6; ++kc) afr[kc] = load_afrag_z(znode, zedge, lane, kc);
    int hi = lane >> 4;
    int nlo = lane & 15;
    for (int nt = 0; nt < 12; ++nt) {
        v8f c = {};
        #pragma unroll
        for (int kc = 0; kc < 6; ++kc) {
            v16h b = load_bfrag(w1f, lane, kc, nt, 12);
            c = __builtin_amdgcn_wmma_f32_16x16x32_f16(false, afr[kc], false, b,
                                                       (short)0, c, false, false);
        }
        float bias = b1[nt * 16 + nlo];
        #pragma unroll
        for (int r = 0; r < 8; ++r)
            h1[(r + 8 * hi) * SZF + nt * 16 + nlo] = c[r] + bias;
    }
    __syncthreads();

    // ---- LayerNorm(192) + ReLU, f16 -> h1n ----
    {
        int m = lane & 15, half_id = lane >> 4;
        const float* rowp = h1 + m * SZF + half_id * 96;
        float s = 0.f, s2 = 0.f;
        #pragma unroll 8
        for (int i = 0; i < 96; ++i) { float x = rowp[i]; s += x; s2 += x * x; }
        s  += __shfl_xor(s, 16, 32);
        s2 += __shfl_xor(s2, 16, 32);
        float mu  = s * (1.f / 192.f);
        float var = s2 * (1.f / 192.f) - mu * mu;
        float rs  = rsqrtf(var + 1e-5f);
        half_t* outp = h1n + m * SZH + half_id * 96;
        const float* lw = lnw + half_id * 96;
        const float* lb = lnb + half_id * 96;
        #pragma unroll 8
        for (int i = 0; i < 96; ++i) {
            float x = (rowp[i] - mu) * rs * lw[i] + lb[i];
            x = x > 0.f ? x : 0.f;
            outp[i] = (half_t)x;
        }
    }
    __syncthreads();

    // ---- GEMM2: out = h1n @ w2 + b2 + ef (residual from LDS zedge) ----
    #pragma unroll
    for (int kc = 0; kc < 6; ++kc) afr[kc] = load_afrag_h(h1n, lane, kc);
    for (int nt = 0; nt < 4; ++nt) {
        v8f c = {};
        #pragma unroll
        for (int kc = 0; kc < 6; ++kc) {
            v16h b = load_bfrag(w2f, lane, kc, nt, 4);
            c = __builtin_amdgcn_wmma_f32_16x16x32_f16(false, afr[kc], false, b,
                                                       (short)0, c, false, false);
        }
        float bias = b2[nt * 16 + nlo];
        int col = nt * 16 + nlo;
        #pragma unroll
        for (int r = 0; r < 8; ++r) {
            long e = e0 + r + 8 * hi;
            if (e < E) {
                float res = zedge[(r + 8 * hi) * SZE + col];
                ef_out[e * 64 + col] = c[r] + bias + res;   // row-wise in-place safe
            }
        }
    }
}

// ================= host side =================
extern "C" void kernel_launch(void* const* d_in, const int* in_sizes, int n_in,
                              void* d_out, int out_size, void* d_ws, size_t ws_size,
                              hipStream_t stream) {
    const float* nf0   = (const float*)d_in[0];
    const float* ef0   = (const float*)d_in[1];
    const int*   ei    = (const int*)d_in[2];
    const float* Wn    = (const float*)d_in[3];    // [L,64,64]
    const float* We    = (const float*)d_in[4];    // [L,64,64]
    const float* atS   = (const float*)d_in[5];    // [L,4,16]
    const float* atD   = (const float*)d_in[6];
    const float* atE   = (const float*)d_in[7];
    const float* gbias = (const float*)d_in[8];    // [L,64]
    const float* lnw   = (const float*)d_in[9];
    const float* lnb   = (const float*)d_in[10];
    const float* w1    = (const float*)d_in[11];   // [L,192,192]
    const float* b1    = (const float*)d_in[12];   // [L,192]
    const float* elnw  = (const float*)d_in[13];
    const float* elnb  = (const float*)d_in[14];
    const float* w2    = (const float*)d_in[15];   // [L,192,64]
    const float* b2    = (const float*)d_in[16];   // [L,64]

    const int N = in_sizes[0] / 64;
    const int E = in_sizes[2] / 2;
    const int L = 4;
    const int* es = ei;
    const int* ed = ei + E;

    float* nf_out = (float*)d_out;                 // [N,64]
    float* ef_out = (float*)d_out + (long)N * 64;  // [E,64]

    // ---- carve workspace ----
    size_t off = 0;
    auto carve = [&](size_t bytes) -> void* {
        void* p = (char*)d_ws + off;
        off += (bytes + 255) & ~(size_t)255;
        return p;
    };
    float*    xh     = (float*)carve((size_t)N * 64 * 4);
    float*    outb   = (float*)carve((size_t)N * 64 * 4);
    float*    a_src  = (float*)carve((size_t)N * 4 * 4);
    float*    a_dst  = (float*)carve((size_t)N * 4 * 4);
    float*    aself  = (float*)carve((size_t)N * 4 * 4);   // alpha_self -> ex_self
    float*    denom  = (float*)carve((size_t)N * 4 * 4);
    unsigned* amax   = (unsigned*)carve((size_t)N * 4 * 4);
    float*    alphaE = (float*)carve((size_t)E * 4 * 4);   // alpha -> ex
    float*    vs     = (float*)carve(64 * 4 * 4);
    float*    vd     = (float*)carve(64 * 4 * 4);
    float*    ve     = (float*)carve(64 * 4 * 4);
    half_t*   w1f    = (half_t*)carve((size_t)DZ * DZ * 2);
    half_t*   w2f    = (half_t*)carve((size_t)DZ * 64 * 2);
    (void)ws_size; (void)n_in; (void)out_size;

    const int TB = 256;
    dim3 blk(TB);
    int gN   = (N + TB - 1) / TB;
    int gN16 = (N * 16 + TB - 1) / TB;
    int gE   = (E + TB - 1) / TB;
    int gE4  = (E * 4 + TB - 1) / TB;
    int tiles = (E + 15) / 16;
    int gEU   = (tiles + EU_WAVES - 1) / EU_WAVES;

    for (int l = 0; l < L; ++l) {
        const float* Wl   = Wn + (long)l * 64 * 64;
        const float* Wel  = We + (long)l * 64 * 64;
        const float* nfin = (l == 0) ? nf0 : nf_out;   // node feats entering layer
        const float* efin = (l == 0) ? ef0 : ef_out;   // edge feats entering layer

        // per-layer attention vectors + f16 fragment-packed MLP weights
        prep_vecs_kernel<<<1, 256, 0, stream>>>(Wl, Wel,
            atS + (long)l * 64, atD + (long)l * 64, atE + (long)l * 64, vs, vd, ve);
        frag_pack_kernel<<<(DZ * DZ + TB - 1) / TB, blk, 0, stream>>>(
            w1 + (long)l * DZ * DZ, w1f, DZ, DZ);
        frag_pack_kernel<<<(DZ * 64 + TB - 1) / TB, blk, 0, stream>>>(
            w2 + (long)l * DZ * 64, w2f, DZ, 64);

        // GAT attention + aggregation
        node_xh_kernel<<<gN16, blk, 0, stream>>>(nfin, Wl, xh, N);
        node_attn_kernel<<<gN, blk, 0, stream>>>(nfin, vs, vd, a_src, a_dst, aself, amax, N);
        edge_alpha_kernel<<<gE, blk, 0, stream>>>(efin, ve, a_src, a_dst, es, ed,
                                                  alphaE, amax, E);
        node_denom_kernel<<<gN, blk, 0, stream>>>(amax, aself, denom, N);
        edge_exp_kernel<<<gE, blk, 0, stream>>>(ed, amax, alphaE, denom, E);
        node_out_init_kernel<<<gN, blk, 0, stream>>>(xh, aself, denom, outb, N);
        edge_scatter_kernel<<<gE4, blk, 0, stream>>>(es, ed, xh, alphaE, denom, outb, E);
        node_finalize_kernel<<<gN, blk, 0, stream>>>(outb, gbias + (long)l * 64,
            lnw + (long)l * 64, lnb + (long)l * 64, nfin, nf_out, N, (l == L - 1) ? 0 : 1);

        // fused edge-update MLP (WMMA + TDM) with residual; in-place on ef_out for l >= 1
        edge_update_kernel<<<gEU, dim3(EU_WAVES * 32), 0, stream>>>(
            nf_out, efin, ef_out, es, ed,
            w1f, b1 + (long)l * DZ,
            elnw + (long)l * DZ, elnb + (long)l * DZ,
            w2f, b2 + (long)l * 64, E);
    }
}
